// VectorQuantizer_1821066134293
// MI455X (gfx1250) — compile-verified, matrix operand order checked by reading, not run on hardware
//
#include <hip/hip_runtime.h>
#include <hip/hip_bf16.h>
#include <float.h>

typedef __attribute__((ext_vector_type(16))) __bf16 v16bf;
typedef __attribute__((ext_vector_type(8)))  float  v8f;

#define N_TOK 32768
#define K_EMB 8192
#define DIM   256
#define DECAY 0.99f
#define EPSV  1e-05f
#define COMMIT 0.25f
#define NTILE (K_EMB / 32)

// ---- CDNA5 async LDS path (guarded; falls back to plain copy) ----
#if defined(__has_builtin)
#if __has_builtin(__builtin_amdgcn_global_load_async_to_lds_b128) && \
    __has_builtin(__builtin_amdgcn_s_wait_asynccnt)
#define HAVE_ASYNC_LDS 1
#endif
#endif

#ifdef HAVE_ASYNC_LDS
typedef int vsi4 __attribute__((vector_size(16)));
typedef __attribute__((address_space(1))) vsi4* gas_v4i;   // global (addrspace 1)
typedef __attribute__((address_space(3))) vsi4* las_v4i;   // LDS (addrspace 3)

__device__ __forceinline__ void async_cp16(const unsigned* g, unsigned* l) {
    __builtin_amdgcn_global_load_async_to_lds_b128(
        (gas_v4i)(g), (las_v4i)(l), 0, 0);
}
#endif

// ---------------- helpers ----------------
__device__ __forceinline__ unsigned f2bf_pack(float a, float b) {
    unsigned ua = __float_as_uint(a);
    unsigned ub = __float_as_uint(b);
    unsigned ra = (ua + 0x7FFFu + ((ua >> 16) & 1u)) >> 16;   // RNE f32->bf16
    unsigned rb = (ub + 0x7FFFu + ((ub >> 16) & 1u)) >> 16;
    return ra | (rb << 16);
}

// ---------------- zero workspace ----------------
__global__ void vq_zero_kernel(float* __restrict__ p, long long n) {
    long long i = (long long)blockIdx.x * blockDim.x + threadIdx.x;
    if (i < n) p[i] = 0.0f;
}

// ---------------- embedding f32 -> packed bf16 ----------------
__global__ void vq_emb2bf16(const float* __restrict__ emb, unsigned* __restrict__ out) {
    const size_t i = (size_t)blockIdx.x * blockDim.x + threadIdx.x;  // K*D/2 items
    const float2 v = *(const float2*)(emb + 2 * i);
    out[i] = f2bf_pack(v.x, v.y);
}

// ---------------- row sum-of-squares (block per row, 256 thr) ----------------
__global__ void vq_rowsq_kernel(const float* __restrict__ x, float* __restrict__ out) {
    const int r = blockIdx.x;
    float v = x[(size_t)r * DIM + threadIdx.x];
    float s = v * v;
    #pragma unroll
    for (int off = 16; off > 0; off >>= 1) s += __shfl_down(s, off, 32);
    __shared__ float partial[8];
    if ((threadIdx.x & 31) == 0) partial[threadIdx.x >> 5] = s;
    __syncthreads();
    if (threadIdx.x == 0) {
        float t = 0.0f;
        #pragma unroll
        for (int i = 0; i < 8; ++i) t += partial[i];
        out[r] = t;
    }
}

// ---------------- fused WMMA distance + argmin ----------------
// 512 blocks x 256 threads (8 waves). Block: 64 z-rows vs all 8192 codes.
// Wave w: mt = w>>1 (16-row tile), nt = w&1 (16-col half of the 32-col tile).
__launch_bounds__(256)
__global__ void vq_argmin_wmma(const float* __restrict__ z,
                               const unsigned* __restrict__ embbf,   // packed bf16 pairs
                               const float* __restrict__ zsq,
                               const float* __restrict__ esq,
                               int*   __restrict__ idx_ws,
                               float* __restrict__ idx_out) {
    // 32KB z-tile + 2 x 16KB embedding double-buffer = 64KB
    __shared__ unsigned smem[64 * 128 + 2 * 32 * 128];
    unsigned* zAu = smem;                 // [64][128] packed bf16
    unsigned* eB  = smem + 64 * 128;      // [2][32][128]

    const int tid  = threadIdx.x;
    const int lane = tid & 31;
    const int w    = tid >> 5;
    const int mt   = w >> 1;
    const int nt   = w & 1;
    const int m    = lane & 15;
    const int half = lane >> 4;
    const int kb2  = half * 4;            // (kbase = 8*half)/2 in u32 units
    const int rowBlock = blockIdx.x * 64;

#ifdef HAVE_ASYNC_LDS
    // kick off tile 0 DMA before doing anything else (4 x b128 per thread)
    {
        const unsigned* g = embbf;        // tile 0 is contiguous 16KB
        #pragma unroll
        for (int j = 0; j < 4; ++j) {
            const int c = tid + 256 * j;  // 16B chunk id
            async_cp16(g + c * 4, eB + c * 4);
        }
    }
#endif

    // stage this block's 64 z-rows, f32 -> bf16, into LDS
    for (int i = tid; i < 64 * 128; i += 256) {
        const int r = i >> 7, p = i & 127;
        const float2 v = *(const float2*)(z + (size_t)(rowBlock + r) * DIM + 2 * p);
        zAu[i] = f2bf_pack(v.x, v.y);
    }
    __syncthreads();

    // preload all 8 A fragments (16x32 bf16) for this wave's row tile
    v16bf aFrag[8];
    #pragma unroll
    for (int d = 0; d < 8; ++d) {
        union { unsigned u[8]; v16bf v; } x;
        const int base = (mt * 16 + m) * 128 + d * 16 + kb2;
        #pragma unroll
        for (int j = 0; j < 4; ++j) { x.u[j] = zAu[base + j]; x.u[4 + j] = zAu[base + 8 + j]; }
        aFrag[d] = x.v;
    }

    float zsqv[8];
    #pragma unroll
    for (int vg = 0; vg < 8; ++vg)
        zsqv[vg] = zsq[rowBlock + mt * 16 + vg + half * 8];

    float minv[8]; int mini[8];
    #pragma unroll
    for (int vg = 0; vg < 8; ++vg) { minv[vg] = FLT_MAX; mini[vg] = 0; }

    const int baseCol = (nt * 16 + m) * 128 + kb2;

    for (int kt = 0; kt < NTILE; ++kt) {
#ifdef HAVE_ASYNC_LDS
        const int cur = kt & 1;
        if (kt + 1 < NTILE) {
            // stream next tile into the other buffer while we compute
            const unsigned* g = embbf + (size_t)(kt + 1) * 4096;
            unsigned* l = eB + (cur ^ 1) * 4096;
            #pragma unroll
            for (int j = 0; j < 4; ++j) {
                const int c = tid + 256 * j;
                async_cp16(g + c * 4, l + c * 4);
            }
            __builtin_amdgcn_s_wait_asynccnt(4);   // current tile's 4 loads done
        } else {
            __builtin_amdgcn_s_wait_asynccnt(0);
        }
        __syncthreads();                           // all waves' DMA landed
#else
        const int cur = 0;
        __syncthreads();
        {
            const uint4* g = (const uint4*)(embbf + (size_t)kt * 4096);
            uint4* l = (uint4*)eB;
            for (int i = tid; i < 1024; i += 256) l[i] = g[i];
        }
        __syncthreads();
#endif
        if (kt + 2 < NTILE)                        // global_prefetch_b8 hint
            __builtin_prefetch(embbf + (size_t)(kt + 2) * 4096 + tid * 4, 0, 1);

        const int colBase = kt * 32;
        const int col  = colBase + nt * 16 + m;
        const float esqv = esq[col];
        const unsigned* eb = eB + cur * 4096;

        // batch-load all 8 B fragments, then run the WMMA chain uninterrupted
        union BF { unsigned u[8]; v16bf v; } b[8];
        #pragma unroll
        for (int d = 0; d < 8; ++d) {
            const int bb = baseCol + d * 16;
            #pragma unroll
            for (int j = 0; j < 4; ++j) { b[d].u[j] = eb[bb + j]; b[d].u[4 + j] = eb[bb + 8 + j]; }
        }

        v8f acc0 = {}, acc1 = {};
        #pragma unroll
        for (int d = 0; d < 8; d += 2) {
            acc0 = __builtin_amdgcn_wmma_f32_16x16x32_bf16(
                       false, aFrag[d],     false, b[d].v,     (short)0, acc0, false, false);
            acc1 = __builtin_amdgcn_wmma_f32_16x16x32_bf16(
                       false, aFrag[d + 1], false, b[d + 1].v, (short)0, acc1, false, false);
        }

        #pragma unroll
        for (int vg = 0; vg < 8; ++vg) {
            const float dist = zsqv[vg] + esqv - 2.0f * (acc0[vg] + acc1[vg]);
            if (dist < minv[vg]) { minv[vg] = dist; mini[vg] = col; }
        }
#ifdef HAVE_ASYNC_LDS
        __syncthreads();   // compute done before next iteration's DMA reuses this buffer
#endif
    }

    // merge scratch aliases the (now dead) z-tile LDS region
    float* sV = (float*)smem;            // [2][64]
    int*   sI = (int*)(smem + 128);      // [2][64]

    // butterfly argmin across the 16 lanes holding each row
    #pragma unroll
    for (int vg = 0; vg < 8; ++vg) {
        float v = minv[vg]; int i = mini[vg];
        #pragma unroll
        for (int off = 1; off < 16; off <<= 1) {
            const float ov = __shfl_xor(v, off, 32);
            const int   oi = __shfl_xor(i, off, 32);
            if (ov < v || (ov == v && oi < i)) { v = ov; i = oi; }
        }
        if (m == 0) {
            const int lr = mt * 16 + vg + half * 8;
            sV[nt * 64 + lr] = v; sI[nt * 64 + lr] = i;
        }
    }
    __syncthreads();
    if (tid < 64) {
        const float v0 = sV[tid], v1 = sV[64 + tid];
        const int   i0 = sI[tid], i1 = sI[64 + tid];
        const int best = (v1 < v0 || (v1 == v0 && i1 < i0)) ? i1 : i0;
        const int row = rowBlock + tid;
        idx_ws[row]  = best;
        idx_out[row] = (float)best;
    }
}

// ---------------- gather z_q, loss partials, scatter counts/dw ----------------
__global__ void vq_scatter(const float* __restrict__ z, const float* __restrict__ emb,
                           const int* __restrict__ idx,
                           float* __restrict__ zq_out,
                           float* __restrict__ dw, float* __restrict__ counts,
                           float* __restrict__ lossAcc) {
    const int r = blockIdx.x;
    const int k = idx[r];
    const int d = threadIdx.x;
    const float zv = z[(size_t)r * DIM + d];
    const float ev = emb[(size_t)k * DIM + d];
    zq_out[(size_t)r * DIM + d] = ev;            // straight-through value == e[idx]
    const float df = ev - zv;
    float s = df * df;
    #pragma unroll
    for (int off = 16; off > 0; off >>= 1) s += __shfl_down(s, off, 32);
    __shared__ float partial[8];
    if ((threadIdx.x & 31) == 0) partial[threadIdx.x >> 5] = s;
    __syncthreads();
    if (threadIdx.x == 0) {
        float t = 0.0f;
        #pragma unroll
        for (int i = 0; i < 8; ++i) t += partial[i];
        atomicAdd(lossAcc, t);
        atomicAdd(&counts[k], 1.0f);
    }
    atomicAdd(&dw[(size_t)k * DIM + d], zv);
}

// ---------------- EMA cluster size + total n ----------------
__global__ void vq_ema_cluster(const float* __restrict__ ema_cs, const float* __restrict__ counts,
                               float* __restrict__ ncs_out, float* __restrict__ nAcc) {
    const int k = blockIdx.x * blockDim.x + threadIdx.x;
    const float v = ema_cs[k] * DECAY + (1.0f - DECAY) * counts[k];
    ncs_out[k] = v;
    float s = v;
    #pragma unroll
    for (int off = 16; off > 0; off >>= 1) s += __shfl_down(s, off, 32);
    __shared__ float partial[8];
    if ((threadIdx.x & 31) == 0) partial[threadIdx.x >> 5] = s;
    __syncthreads();
    if (threadIdx.x == 0) {
        float t = 0.0f;
        #pragma unroll
        for (int i = 0; i < 8; ++i) t += partial[i];
        atomicAdd(nAcc, t);
    }
}

// ---------------- EMA codebook weights ----------------
__global__ void vq_ema_w(const float* __restrict__ ema_w, const float* __restrict__ dw,
                         float* __restrict__ out) {
    const size_t i = (size_t)blockIdx.x * blockDim.x + threadIdx.x;
    out[i] = ema_w[i] * DECAY + (1.0f - DECAY) * dw[i];
}

// ---------------- normalized embedding + loss finalize ----------------
__global__ void vq_finalize(const float* __restrict__ ncs, const float* __restrict__ emaw,
                            const float* __restrict__ nAcc, const float* __restrict__ lossAcc,
                            float* __restrict__ emb_out, float* __restrict__ loss_out) {
    const size_t i = (size_t)blockIdx.x * blockDim.x + threadIdx.x;
    const int k = (int)(i >> 8);
    const float n = *nAcc;
    const float cs = (ncs[k] + EPSV) / (n + (float)K_EMB * EPSV) * n;
    emb_out[i] = emaw[i] / cs;
    if (i == 0)
        *loss_out = COMMIT * (*lossAcc) / (float)((size_t)N_TOK * DIM);
}

// ---------------- launch ----------------
extern "C" void kernel_launch(void* const* d_in, const int* in_sizes, int n_in,
                              void* d_out, int out_size, void* d_ws, size_t ws_size,
                              hipStream_t stream) {
    (void)in_sizes; (void)n_in; (void)out_size; (void)ws_size;
    const float* z      = (const float*)d_in[0];   // [32,1024,256]
    const float* emb    = (const float*)d_in[1];   // [8192,256]
    const float* ema_cs = (const float*)d_in[2];   // [8192]
    const float* ema_w  = (const float*)d_in[3];   // [8192,256]
    float* out = (float*)d_out;
    float* ws  = (float*)d_ws;

    // workspace layout (float offsets)
    const size_t WS_ESQ    = 0;                    // 8192
    const size_t WS_ZSQ    = 8192;                 // 32768
    const size_t WS_COUNTS = 40960;                // 8192
    const size_t WS_DW     = 49152;                // 2097152
    const size_t WS_IDX    = 2146304;              // 32768 (int)
    const size_t WS_LOSS   = 2179072;              // 1
    const size_t WS_NACC   = 2179073;              // 1
    const size_t WS_EBF    = 2179076;              // 1048576 u32 (bf16 codebook)

    // output layout (float offsets, reference return order)
    const size_t OUT_ZQ   = 0;                     // 8388608
    const size_t OUT_LOSS = 8388608;               // 1
    const size_t OUT_IDX  = 8388609;               // 32768
    const size_t OUT_EMB  = 8421377;               // 2097152
    const size_t OUT_NCS  = 10518529;              // 8192
    const size_t OUT_EMAW = 10526721;              // 2097152

    const long long nz = (long long)(WS_NACC + 1 - WS_COUNTS);
    vq_zero_kernel<<<(unsigned)((nz + 255) / 256), 256, 0, stream>>>(ws + WS_COUNTS, nz);

    vq_rowsq_kernel<<<N_TOK, 256, 0, stream>>>(z, ws + WS_ZSQ);
    vq_rowsq_kernel<<<K_EMB, 256, 0, stream>>>(emb, ws + WS_ESQ);
    vq_emb2bf16<<<(K_EMB * DIM / 2) / 256, 256, 0, stream>>>(emb, (unsigned*)(ws + WS_EBF));

    vq_argmin_wmma<<<N_TOK / 64, 256, 0, stream>>>(z, (const unsigned*)(ws + WS_EBF),
                                                   ws + WS_ZSQ, ws + WS_ESQ,
                                                   (int*)(ws + WS_IDX), out + OUT_IDX);

    vq_scatter<<<N_TOK, 256, 0, stream>>>(z, emb, (const int*)(ws + WS_IDX),
                                          out + OUT_ZQ, ws + WS_DW, ws + WS_COUNTS,
                                          ws + WS_LOSS);

    vq_ema_cluster<<<K_EMB / 256, 256, 0, stream>>>(ema_cs, ws + WS_COUNTS,
                                                    out + OUT_NCS, ws + WS_NACC);

    vq_ema_w<<<(K_EMB * DIM) / 256, 256, 0, stream>>>(ema_w, ws + WS_DW, out + OUT_EMAW);

    vq_finalize<<<(K_EMB * DIM) / 256, 256, 0, stream>>>(out + OUT_NCS, out + OUT_EMAW,
                                                         ws + WS_NACC, ws + WS_LOSS,
                                                         out + OUT_EMB, out + OUT_LOSS);
}
